// FLayer_33019708572201
// MI455X (gfx1250) — compile-verified, matrix operand order checked by reading, not run on hardware
//
#include <hip/hip_runtime.h>

// ---------------------------------------------------------------------------
// Matching-network read head: 8 chained LSTM+attention steps.
// B=32, T=128 -> BT=4096 rows; E=1024; gates 4E=4096; support N*k=320/batch.
// Strategy: bf16 WMMA (v_wmma_f32_16x16x32_bf16) for all GEMMs, f32 state.
//   - x@W_ih^T and h@W_hh^T fused into one K=2048 GEMM via [x|h] and [Wih;Whh]
//   - LSTM nonlinearity fused into GEMM epilogue (gates never hit memory)
//   - global->LDS staging uses CDNA5 async copies (ASYNCcnt) with
//     double-buffered LDS tiles: next K-tile streams in during WMMA compute
// ---------------------------------------------------------------------------

typedef __attribute__((ext_vector_type(16))) __bf16 v16bf;
typedef __attribute__((ext_vector_type(8)))  __bf16 v8bf;
typedef __attribute__((ext_vector_type(8)))  float  v8f;

#define BT      4096   // B*T rows
#define EE      1024   // encoding size
#define KK2     2048   // concat K = E(x) + E(h)
#define NS      320    // N*k support rows per batch
#define NBATCH  32
#define KSTEPS  8

// ---- CDNA5 async global->LDS copy (16B per lane, tracked by ASYNCcnt) -----
// LDS aperture maps addr[31:0] to the wave's LDS offset (ISA 10.2), so
// truncating the generic pointer yields the VDST LDS address.
__device__ __forceinline__ void async_b128(const void* lds_dst, const void* gsrc) {
  asm volatile("global_load_async_to_lds_b128 %0, %1, off"
               :: "v"((unsigned int)(unsigned long long)(uintptr_t)lds_dst),
                  "v"((unsigned long long)(uintptr_t)gsrc)
               : "memory");
}
__device__ __forceinline__ void wait_async0() {
#if __has_builtin(__builtin_amdgcn_s_wait_asynccnt)
  __builtin_amdgcn_s_wait_asynccnt(0);
#else
  asm volatile("s_wait_asynccnt 0x0" ::: "memory");
#endif
}

// ---- WMMA fragment load from LDS tile stored as [row][32] bf16 -------------
// 16-bit A layout: lanes 0-15 -> M=lane,  K = {0..7, 16..23}
//                  lanes 16-31-> M=l-16,  K = {8..15, 24..31}
// B (32x16) mirrors this with N when the tile is stored column-major [n][k].
__device__ __forceinline__ v16bf frag16(const __bf16* rowptr, int sel) {
  const v8bf* p = (const v8bf*)rowptr;
  v8bf lo = p[sel];       // K = sel*8 .. sel*8+7
  v8bf hi = p[2 + sel];   // K = 16+sel*8 ..
  return __builtin_shufflevector(lo, hi, 0,1,2,3,4,5,6,7,8,9,10,11,12,13,14,15);
}

__device__ __forceinline__ float sigmf(float x) { return 1.0f / (1.0f + __expf(-x)); }

// ---------------------------------------------------------------------------
// Prep kernels
// ---------------------------------------------------------------------------
__global__ void prep_wcomb(const float* __restrict__ Wih, const float* __restrict__ Whh,
                           __bf16* __restrict__ Wc) {
  for (size_t i = blockIdx.x * blockDim.x + threadIdx.x; i < (size_t)4096 * KK2;
       i += (size_t)gridDim.x * blockDim.x) {
    size_t row = i >> 11;
    int    k   = (int)(i & 2047);
    float v = (k < EE) ? Wih[row * EE + k] : Whh[row * EE + (k - EE)];
    Wc[i] = (__bf16)v;
  }
}

__global__ void prep_bias(const float* __restrict__ bih, const float* __restrict__ bhh,
                          float* __restrict__ bsum) {
  int i = blockIdx.x * blockDim.x + threadIdx.x;
  if (i < 4096) bsum[i] = bih[i] + bhh[i];
}

__global__ void prep_state(const float* __restrict__ x, __bf16* __restrict__ xh,
                           float* __restrict__ h, float* __restrict__ c,
                           float* __restrict__ r) {
  for (size_t i = blockIdx.x * blockDim.x + threadIdx.x; i < (size_t)BT * EE;
       i += (size_t)gridDim.x * blockDim.x) {
    size_t row = i >> 10;
    int    col = (int)(i & 1023);
    xh[row * KK2 + col]      = (__bf16)x[i];   // x half (static)
    xh[row * KK2 + EE + col] = (__bf16)0.0f;   // h half (rewritten each step)
    h[i] = 0.0f; c[i] = 0.0f; r[i] = 0.0f;
  }
}

__global__ void prep_support(const float* __restrict__ sup, __bf16* __restrict__ sbf,
                             __bf16* __restrict__ sTbf) {
  // sup: [B][320][1024]; sTbf: [B][1024][320]
  for (size_t i = blockIdx.x * blockDim.x + threadIdx.x; i < (size_t)NBATCH * NS * EE;
       i += (size_t)gridDim.x * blockDim.x) {
    size_t b  = i / ((size_t)NS * EE);
    size_t re = i - b * (size_t)NS * EE;
    int    j  = (int)(re >> 10);
    int    e  = (int)(re & 1023);
    __bf16 v = (__bf16)sup[i];
    sbf[i] = v;
    sTbf[(b * EE + e) * NS + j] = v;
  }
}

// h_read = h + r  -> bf16 into xh h-half (this is both GEMM input and q)
__global__ void read_add(const float* __restrict__ h, const float* __restrict__ r,
                         __bf16* __restrict__ xh) {
  for (size_t i = blockIdx.x * blockDim.x + threadIdx.x; i < (size_t)BT * EE;
       i += (size_t)gridDim.x * blockDim.x) {
    size_t row = i >> 10;
    int    col = (int)(i & 1023);
    xh[row * KK2 + EE + col] = (__bf16)(h[i] + r[i]);
  }
}

// ---------------------------------------------------------------------------
// Fused gate GEMM + LSTM pointwise.
// grid = (BT/128, EE/64), block = 256 (8 waves). Wave w owns rows [w*16,w*16+16).
// Each wave accumulates 4 gates x 4 n-subtiles (16 cols) at e, e+1024, ...
// ---------------------------------------------------------------------------
__global__ __launch_bounds__(256) void gates_lstm_kernel(
    const __bf16* __restrict__ xh, const __bf16* __restrict__ Wc,
    const float* __restrict__ bsum, const float* __restrict__ xf,
    float* __restrict__ hbuf, float* __restrict__ cbuf) {
  __shared__ __bf16 lsA[2][128][32];   // [buf][m][k]
  __shared__ __bf16 lsB[2][256][32];   // [buf][n][k], n = gate*64 + e_local

  const int tid  = threadIdx.x;
  const int lane = tid & 31, wave = tid >> 5;
  const int sel  = lane >> 4, half = lane & 15;
  const int rowbase = blockIdx.x * 128;
  const int ebase   = blockIdx.y * 64;

  auto stage = [&](int p, int kb) {
#pragma unroll
    for (int ch = tid; ch < 512; ch += 256) {            // A: 128x32 bf16
      int rr = ch >> 2, k8 = (ch & 3) << 3;
      async_b128(&lsA[p][rr][k8], xh + (size_t)(rowbase + rr) * KK2 + kb + k8);
    }
#pragma unroll
    for (int ch = tid; ch < 1024; ch += 256) {           // B: 256x32 bf16
      int col = ch >> 2, k8 = (ch & 3) << 3;
      int g = col >> 6, e = ebase + (col & 63);
      async_b128(&lsB[p][col][k8], Wc + (size_t)(g * EE + e) * KK2 + kb + k8);
    }
  };

  v8f acc[4][4];
  const v8f vzero = {0.f,0.f,0.f,0.f,0.f,0.f,0.f,0.f};
#pragma unroll
  for (int g = 0; g < 4; ++g)
#pragma unroll
    for (int s = 0; s < 4; ++s) acc[g][s] = vzero;

  stage(0, 0);
  int p = 0;
  for (int kb = 0; kb < KK2; kb += 32) {
    wait_async0();            // my tile-i async copies have landed in LDS
    __syncthreads();          // everyone's landed; prior compute on other buf done
    if (kb + 32 < KK2) stage(p ^ 1, kb + 32);   // stream next tile during compute

    v16bf a = frag16(&lsA[p][wave * 16 + half][0], sel);
#pragma unroll
    for (int g = 0; g < 4; ++g)
#pragma unroll
      for (int s = 0; s < 4; ++s) {
        v16bf b = frag16(&lsB[p][g * 64 + s * 16 + half][0], sel);
        acc[g][s] = __builtin_amdgcn_wmma_f32_16x16x32_bf16(
            false, a, false, b, (short)0, acc[g][s], false, false);
      }
    p ^= 1;
  }

  // Epilogue: C tile layout -> vgpr r: M = r + 8*sel, N = half
#pragma unroll
  for (int s = 0; s < 4; ++s) {
    int e = ebase + s * 16 + half;
    float bi = bsum[e], bf_ = bsum[EE + e], bg = bsum[2 * EE + e], bo = bsum[3 * EE + e];
#pragma unroll
    for (int rr = 0; rr < 8; ++rr) {
      int row = rowbase + wave * 16 + rr + 8 * sel;
      float iv = sigmf(acc[0][s][rr] + bi);
      float fv = sigmf(acc[1][s][rr] + bf_);
      float gv = tanhf(acc[2][s][rr] + bg);
      float ov = sigmf(acc[3][s][rr] + bo);
      size_t idx = (size_t)row * EE + e;
      float cn = fv * cbuf[idx] + iv * gv;
      cbuf[idx] = cn;
      hbuf[idx] = ov * tanhf(cn) + xf[idx];   // skip connection
    }
  }
}

// ---------------------------------------------------------------------------
// dots[b,t,j] = q . s_j   (q = xh h-half). grid = (32 batches, 5 j-tiles)
// ---------------------------------------------------------------------------
__global__ __launch_bounds__(256) void dots_kernel(
    const __bf16* __restrict__ xh, const __bf16* __restrict__ sbf,
    float* __restrict__ dots) {
  __shared__ __bf16 lsA[2][128][32];
  __shared__ __bf16 lsB[2][64][32];
  const int tid = threadIdx.x;
  const int lane = tid & 31, wave = tid >> 5;
  const int sel = lane >> 4, half = lane & 15;
  const int b = blockIdx.x;
  const int rowbase = b * 128;
  const int jbase = blockIdx.y * 64;

  auto stage = [&](int p, int kb) {
#pragma unroll
    for (int ch = tid; ch < 512; ch += 256) {
      int rr = ch >> 2, k8 = (ch & 3) << 3;
      async_b128(&lsA[p][rr][k8], xh + (size_t)(rowbase + rr) * KK2 + EE + kb + k8);
    }
    {
      int col = tid >> 2, k8 = (tid & 3) << 3;           // 256 chunks exactly
      async_b128(&lsB[p][col][k8], sbf + ((size_t)b * NS + jbase + col) * EE + kb + k8);
    }
  };

  v8f acc[4];
  const v8f vzero = {0.f,0.f,0.f,0.f,0.f,0.f,0.f,0.f};
#pragma unroll
  for (int s = 0; s < 4; ++s) acc[s] = vzero;

  stage(0, 0);
  int p = 0;
  for (int kb = 0; kb < EE; kb += 32) {
    wait_async0();
    __syncthreads();
    if (kb + 32 < EE) stage(p ^ 1, kb + 32);

    v16bf a = frag16(&lsA[p][wave * 16 + half][0], sel);
#pragma unroll
    for (int s = 0; s < 4; ++s) {
      v16bf bb = frag16(&lsB[p][s * 16 + half][0], sel);
      acc[s] = __builtin_amdgcn_wmma_f32_16x16x32_bf16(
          false, a, false, bb, (short)0, acc[s], false, false);
    }
    p ^= 1;
  }
#pragma unroll
  for (int s = 0; s < 4; ++s)
#pragma unroll
    for (int rr = 0; rr < 8; ++rr) {
      int row = rowbase + wave * 16 + rr + 8 * sel;
      dots[(size_t)row * NS + jbase + s * 16 + half] = acc[s][rr];
    }
}

// softmax over 320 per row; att -> bf16
__global__ __launch_bounds__(512) void softmax_kernel(
    const float* __restrict__ dots, __bf16* __restrict__ att) {
  __shared__ float red[512];
  const int row = blockIdx.x;
  const int t = threadIdx.x;
  float v = (t < NS) ? dots[(size_t)row * NS + t] : -3.0e38f;
  red[t] = v;
  __syncthreads();
  for (int s = 256; s > 0; s >>= 1) {
    if (t < s) red[t] = fmaxf(red[t], red[t + s]);
    __syncthreads();
  }
  float m = red[0];
  __syncthreads();
  float ex = (t < NS) ? __expf(v - m) : 0.0f;
  red[t] = ex;
  __syncthreads();
  for (int s = 256; s > 0; s >>= 1) {
    if (t < s) red[t] += red[t + s];
    __syncthreads();
  }
  float inv = 1.0f / red[0];
  if (t < NS) att[(size_t)row * NS + t] = (__bf16)(ex * inv);
}

// r[b,t,e] = att @ S : A = att[4096][320] bf16, B = sT[b][e][320].
// grid = (32 batches, 16 e-tiles)
__global__ __launch_bounds__(256) void rmat_kernel(
    const __bf16* __restrict__ att, const __bf16* __restrict__ sT,
    float* __restrict__ rbuf) {
  __shared__ __bf16 lsA[2][128][32];
  __shared__ __bf16 lsB[2][64][32];
  const int tid = threadIdx.x;
  const int lane = tid & 31, wave = tid >> 5;
  const int sel = lane >> 4, half = lane & 15;
  const int b = blockIdx.x;
  const int rowbase = b * 128;
  const int ebase = blockIdx.y * 64;

  auto stage = [&](int p, int kb) {
#pragma unroll
    for (int ch = tid; ch < 512; ch += 256) {
      int rr = ch >> 2, k8 = (ch & 3) << 3;
      async_b128(&lsA[p][rr][k8], att + (size_t)(rowbase + rr) * NS + kb + k8);
    }
    {
      int col = tid >> 2, k8 = (tid & 3) << 3;
      async_b128(&lsB[p][col][k8], sT + ((size_t)b * EE + ebase + col) * NS + kb + k8);
    }
  };

  v8f acc[4];
  const v8f vzero = {0.f,0.f,0.f,0.f,0.f,0.f,0.f,0.f};
#pragma unroll
  for (int s = 0; s < 4; ++s) acc[s] = vzero;

  stage(0, 0);
  int p = 0;
  for (int kb = 0; kb < NS; kb += 32) {
    wait_async0();
    __syncthreads();
    if (kb + 32 < NS) stage(p ^ 1, kb + 32);

    v16bf a = frag16(&lsA[p][wave * 16 + half][0], sel);
#pragma unroll
    for (int s = 0; s < 4; ++s) {
      v16bf bb = frag16(&lsB[p][s * 16 + half][0], sel);
      acc[s] = __builtin_amdgcn_wmma_f32_16x16x32_bf16(
          false, a, false, bb, (short)0, acc[s], false, false);
    }
    p ^= 1;
  }
#pragma unroll
  for (int s = 0; s < 4; ++s)
#pragma unroll
    for (int rr = 0; rr < 8; ++rr) {
      int row = rowbase + wave * 16 + rr + 8 * sel;
      rbuf[(size_t)row * EE + ebase + s * 16 + half] = acc[s][rr];
    }
}

// ---------------------------------------------------------------------------
extern "C" void kernel_launch(void* const* d_in, const int* in_sizes, int n_in,
                              void* d_out, int out_size, void* d_ws, size_t ws_size,
                              hipStream_t stream) {
  (void)in_sizes; (void)n_in; (void)out_size; (void)ws_size;
  const float* x    = (const float*)d_in[0];  // [32,128,1024]
  const float* sup  = (const float*)d_in[1];  // [32,64,5,1024]
  const float* Wih  = (const float*)d_in[2];  // [4096,1024]
  const float* Whh  = (const float*)d_in[3];  // [4096,1024]
  const float* bih  = (const float*)d_in[4];  // [4096]
  const float* bhh  = (const float*)d_in[5];  // [4096]

  char* ws = (char*)d_ws;
  size_t off = 0;
  auto take = [&](size_t bytes) { char* p = ws + off; off += (bytes + 255) & ~(size_t)255; return p; };

  __bf16* Wc   = (__bf16*)take((size_t)4096 * KK2 * 2);      // 16 MB
  __bf16* xh   = (__bf16*)take((size_t)BT * KK2 * 2);        // 16 MB
  __bf16* sbf  = (__bf16*)take((size_t)NBATCH * NS * EE * 2);// 20 MB
  __bf16* sTbf = (__bf16*)take((size_t)NBATCH * EE * NS * 2);// 20 MB
  float*  hbuf = (float*)take((size_t)BT * EE * 4);          // 16 MB
  float*  cbuf = (float*)take((size_t)BT * EE * 4);          // 16 MB
  float*  rbuf = (float*)take((size_t)BT * EE * 4);          // 16 MB
  float*  dots = (float*)take((size_t)BT * NS * 4);          // 5 MB
  __bf16* att  = (__bf16*)take((size_t)BT * NS * 2);         // 2.5 MB
  float*  bsum = (float*)take((size_t)4096 * 4);

  // ---- prep ----
  prep_wcomb  <<<2048, 256, 0, stream>>>(Wih, Whh, Wc);
  prep_bias   <<<16,   256, 0, stream>>>(bih, bhh, bsum);
  prep_state  <<<4096, 256, 0, stream>>>(x, xh, hbuf, cbuf, rbuf);
  prep_support<<<4096, 256, 0, stream>>>(sup, sbf, sTbf);

  dim3 gGates(BT / 128, EE / 64);    // 32 x 16
  dim3 gDots(NBATCH, NS / 64);       // 32 x 5
  dim3 gR(NBATCH, EE / 64);          // 32 x 16

  for (int step = 0; step < KSTEPS; ++step) {
    read_add<<<4096, 256, 0, stream>>>(hbuf, rbuf, xh);
    gates_lstm_kernel<<<gGates, 256, 0, stream>>>(xh, Wc, bsum, x, hbuf, cbuf);
    if (step < KSTEPS - 1) {  // last step's r is dead
      dots_kernel   <<<gDots, 256, 0, stream>>>(xh, sbf, dots);
      softmax_kernel<<<BT, 512, 0, stream>>>(dots, att);
      rmat_kernel   <<<gR, 256, 0, stream>>>(att, sTbf, rbuf);
    }
  }

  hipMemcpyAsync(d_out, hbuf, (size_t)BT * EE * 4, hipMemcpyDeviceToDevice, stream);
}
